// Seq2Seq_40458591929165
// MI455X (gfx1250) — matrix operand order repaired and checked by reading, not compiled, not run on hardware
//
#include <hip/hip_runtime.h>

// ---------------------------------------------------------------------------
// Types for CDNA5 WMMA (wave32): v_wmma_f32_16x16x32_bf16
// ---------------------------------------------------------------------------
typedef __attribute__((ext_vector_type(16))) __bf16 v16bf;
typedef __attribute__((ext_vector_type(8)))  __bf16 v8bf;
typedef __attribute__((ext_vector_type(8)))  float  v8f;

// Problem sizes (fixed by the reference)
#define SQ 64
#define TQ 64
#define BB 32
#define HH 1024
#define VV 32000

__device__ __forceinline__ __bf16 to_bf16(float f) {
  unsigned u = __builtin_bit_cast(unsigned, f);
  u += 0x7FFFu + ((u >> 16) & 1u);              // round-to-nearest-even
  unsigned short h = (unsigned short)(u >> 16);
  return __builtin_bit_cast(__bf16, h);
}

#define WMMA_BF16(a, b, c) \
  __builtin_amdgcn_wmma_f32_16x16x32_bf16(false, (a), false, (b), (short)0, (c), false, false)

// A fragment: 16x32 bf16 tile of row-major A (stride K); pa already includes
// row*K and the lane-half K offset (0 or 16). Elements 0..15 <- pa[k..k+15]
// (contiguous K order, matching B).
__device__ __forceinline__ v16bf load_a(const __bf16* pa, int k) {
  v8bf lo = *(const v8bf*)(pa + k);
  v8bf hi = *(const v8bf*)(pa + k + 8);
  v16bf a;
#pragma unroll
  for (int i = 0; i < 8; ++i) { a[i] = lo[i]; a[i + 8] = hi[i]; }
  return a;
}

// ---------------------------------------------------------------------------
// Single-input GEMM: C[32,N] = A[32,K] * W[N,K]^T + bias.
// 4 N-tiles per wave (A fragment reused 4x), 2-deep K software pipeline.
// ---------------------------------------------------------------------------
template <bool RELU_BF16>
__device__ __forceinline__ void
gemm4_body(const __bf16* __restrict__ A, const __bf16* __restrict__ B,
           const float* __restrict__ bias, float* __restrict__ Cf,
           __bf16* __restrict__ Cb, int N, int K) {
  int lane = threadIdx.x & 31;
  int wave = threadIdx.x >> 5;
  int ng = N >> 6;                       // groups of 4 tiles along N
  int grp = blockIdx.x * 8 + wave;
  if (grp >= 2 * ng) return;             // wave-uniform -> EXEC stays all ones
  int m0 = (grp / ng) << 4;
  int n0 = (grp % ng) << 6;
  int l15 = lane & 15;
  const __bf16* pa = A + (size_t)(m0 + l15) * K + ((lane & 16) ? 16 : 0);
  const __bf16* pb0 = B + (size_t)(n0 + 0 * 16 + l15) * K + ((lane & 16) ? 16 : 0);
  const __bf16* pb1 = B + (size_t)(n0 + 1 * 16 + l15) * K + ((lane & 16) ? 16 : 0);
  const __bf16* pb2 = B + (size_t)(n0 + 2 * 16 + l15) * K + ((lane & 16) ? 16 : 0);
  const __bf16* pb3 = B + (size_t)(n0 + 3 * 16 + l15) * K + ((lane & 16) ? 16 : 0);

  v8f acc0 = {}, acc1 = {}, acc2 = {}, acc3 = {};
  v16bf a0, a1, b00, b01, b02, b03, b10, b11, b12, b13;
  a0  = load_a(pa, 0);
  b00 = *(const v16bf*)(pb0);
  b01 = *(const v16bf*)(pb1);
  b02 = *(const v16bf*)(pb2);
  b03 = *(const v16bf*)(pb3);
  for (int k0 = 0; k0 < K; k0 += 64) {       // K is a multiple of 64
    a1  = load_a(pa, k0 + 32);
    b10 = *(const v16bf*)(pb0 + k0 + 32);
    b11 = *(const v16bf*)(pb1 + k0 + 32);
    b12 = *(const v16bf*)(pb2 + k0 + 32);
    b13 = *(const v16bf*)(pb3 + k0 + 32);
    acc0 = WMMA_BF16(a0, b00, acc0);
    acc1 = WMMA_BF16(a0, b01, acc1);
    acc2 = WMMA_BF16(a0, b02, acc2);
    acc3 = WMMA_BF16(a0, b03, acc3);
    if (k0 + 64 < K) {
      a0  = load_a(pa, k0 + 64);
      b00 = *(const v16bf*)(pb0 + k0 + 64);
      b01 = *(const v16bf*)(pb1 + k0 + 64);
      b02 = *(const v16bf*)(pb2 + k0 + 64);
      b03 = *(const v16bf*)(pb3 + k0 + 64);
    }
    acc0 = WMMA_BF16(a1, b10, acc0);
    acc1 = WMMA_BF16(a1, b11, acc1);
    acc2 = WMMA_BF16(a1, b12, acc2);
    acc3 = WMMA_BF16(a1, b13, acc3);
  }
  int rbase = m0 + ((lane & 16) ? 8 : 0);    // C layout: v_r -> M = r (+8 hi half)
  v8f* accs[4] = {&acc0, &acc1, &acc2, &acc3};
#pragma unroll
  for (int t = 0; t < 4; ++t) {
    int col = n0 + t * 16 + l15;
    float bv = bias[col];
#pragma unroll
    for (int r = 0; r < 8; ++r) {
      float v = (*accs[t])[r] + bv;
      if (RELU_BF16) Cb[(size_t)(rbase + r) * N + col] = to_bf16(fmaxf(v, 0.0f));
      else           Cf[(size_t)(rbase + r) * N + col] = v;
    }
  }
}

__global__ void __launch_bounds__(256)
wmma_gemm4(const __bf16* __restrict__ A, const __bf16* __restrict__ B,
           const float* __restrict__ bias, float* __restrict__ C, int N, int K) {
  gemm4_body<false>(A, B, bias, C, nullptr, N, K);
}

__global__ void __launch_bounds__(256)
wmma_gemm4_relu_bf16(const __bf16* __restrict__ A, const __bf16* __restrict__ B,
                     const float* __restrict__ bias, __bf16* __restrict__ C,
                     int N, int K) {
  gemm4_body<true>(A, B, bias, nullptr, C, N, K);
}

// ---------------------------------------------------------------------------
// Dual-input LSTM GEMM: g[32,4H] = x*Wih^T + h*Whh^T + b.
// 2 N-tiles per wave, branch-free inner loop, 2-deep K pipeline.
// ---------------------------------------------------------------------------
__global__ void __launch_bounds__(256)
wmma_gemm_dual2(const __bf16* __restrict__ Ax, const __bf16* __restrict__ Bx,
                const __bf16* __restrict__ Ah, const __bf16* __restrict__ Bh,
                const float* __restrict__ bias, float* __restrict__ C,
                int N, int K) {
  int lane = threadIdx.x & 31;
  int wave = threadIdx.x >> 5;
  int ng = N >> 5;                       // groups of 2 tiles along N
  int grp = blockIdx.x * 8 + wave;
  if (grp >= 2 * ng) return;
  int m0 = (grp / ng) << 4;
  int n0 = (grp % ng) << 5;
  int l15 = lane & 15;
  int koffA = (lane & 16) ? 16 : 0;
  int koffB = (lane & 16) ? 16 : 0;
  const __bf16* pax = Ax + (size_t)(m0 + l15) * K + koffA;
  const __bf16* pah = Ah + (size_t)(m0 + l15) * K + koffA;
  const __bf16* pbx0 = Bx + (size_t)(n0 + l15) * K + koffB;
  const __bf16* pbx1 = Bx + (size_t)(n0 + 16 + l15) * K + koffB;
  const __bf16* pbh0 = Bh + (size_t)(n0 + l15) * K + koffB;
  const __bf16* pbh1 = Bh + (size_t)(n0 + 16 + l15) * K + koffB;

  v8f acc0 = {}, acc1 = {};
  v16bf ax0, ah0, bx00, bx01, bh00, bh01;
  v16bf ax1, ah1, bx10, bx11, bh10, bh11;
  ax0  = load_a(pax, 0);
  ah0  = load_a(pah, 0);
  bx00 = *(const v16bf*)(pbx0);
  bx01 = *(const v16bf*)(pbx1);
  bh00 = *(const v16bf*)(pbh0);
  bh01 = *(const v16bf*)(pbh1);
  for (int k0 = 0; k0 < K; k0 += 64) {
    ax1  = load_a(pax, k0 + 32);
    ah1  = load_a(pah, k0 + 32);
    bx10 = *(const v16bf*)(pbx0 + k0 + 32);
    bx11 = *(const v16bf*)(pbx1 + k0 + 32);
    bh10 = *(const v16bf*)(pbh0 + k0 + 32);
    bh11 = *(const v16bf*)(pbh1 + k0 + 32);
    acc0 = WMMA_BF16(ax0, bx00, acc0);
    acc1 = WMMA_BF16(ax0, bx01, acc1);
    acc0 = WMMA_BF16(ah0, bh00, acc0);
    acc1 = WMMA_BF16(ah0, bh01, acc1);
    if (k0 + 64 < K) {
      ax0  = load_a(pax, k0 + 64);
      ah0  = load_a(pah, k0 + 64);
      bx00 = *(const v16bf*)(pbx0 + k0 + 64);
      bx01 = *(const v16bf*)(pbx1 + k0 + 64);
      bh00 = *(const v16bf*)(pbh0 + k0 + 64);
      bh01 = *(const v16bf*)(pbh1 + k0 + 64);
    }
    acc0 = WMMA_BF16(ax1, bx10, acc0);
    acc1 = WMMA_BF16(ax1, bx11, acc1);
    acc0 = WMMA_BF16(ah1, bh10, acc0);
    acc1 = WMMA_BF16(ah1, bh11, acc1);
  }
  int rbase = m0 + ((lane & 16) ? 8 : 0);
  {
    int col = n0 + l15;
    float bv = bias[col];
#pragma unroll
    for (int r = 0; r < 8; ++r)
      C[(size_t)(rbase + r) * N + col] = acc0[r] + bv;
  }
  {
    int col = n0 + 16 + l15;
    float bv = bias[col];
#pragma unroll
    for (int r = 0; r < 8; ++r)
      C[(size_t)(rbase + r) * N + col] = acc1[r] + bv;
  }
}

// ---------------------------------------------------------------------------
// Small supporting kernels
// ---------------------------------------------------------------------------
__global__ void __launch_bounds__(256)
cvt_bf16(const float* __restrict__ src, __bf16* __restrict__ dst, int n) {
  int i4 = (blockIdx.x * 256 + threadIdx.x) * 4;
  if (i4 >= n) return;
  float4 v = *(const float4*)(src + i4);
  dst[i4 + 0] = to_bf16(v.x);
  dst[i4 + 1] = to_bf16(v.y);
  dst[i4 + 2] = to_bf16(v.z);
  dst[i4 + 3] = to_bf16(v.w);
}

__global__ void __launch_bounds__(256)
init_state(float* h, float* c, __bf16* h_bf, int* prev_tok, float* loss) {
  int idx = blockIdx.x * 256 + threadIdx.x;   // 32768 threads
  h[idx] = 0.0f;
  c[idx] = 0.0f;
  h_bf[idx] = to_bf16(0.0f);
  if (idx < BB) prev_tok[idx] = 0;
  if (idx == 0) loss[0] = 0.0f;
}

// encoder input prep: x_bf = bf16(enc_emb[input[s,b]])
__global__ void __launch_bounds__(256)
enc_prep(const int* __restrict__ input, const float* __restrict__ emb,
         __bf16* __restrict__ x_bf, int step) {
  int b = blockIdx.x;
  int tok = input[step * BB + b];
  const float* er = emb + (size_t)tok * HH;
  for (int j = threadIdx.x; j < HH; j += 256)
    x_bf[b * HH + j] = to_bf16(er[j]);
}

// decoder input prep: token select, x = relu(dec_emb[tok]), cat1=[x|h], cat2[:H]=x
__global__ void __launch_bounds__(256)
dec_prep(const int* __restrict__ target, const int* __restrict__ use_tf_p,
         const int* __restrict__ prev_tok, const float* __restrict__ emb,
         const float* __restrict__ h, float* __restrict__ x_f32,
         __bf16* __restrict__ cat1, __bf16* __restrict__ cat2, int step) {
  int b = blockIdx.x;
  int teacher = (step == 0) ? 0 : target[(step - 1) * BB + b];
  int tok = use_tf_p[0] ? teacher : prev_tok[b];
  const float* er = emb + (size_t)tok * HH;
  for (int j = threadIdx.x; j < HH; j += 256) {
    float x = fmaxf(er[j], 0.0f);
    x_f32[b * HH + j] = x;
    __bf16 xb = to_bf16(x);
    cat1[b * 2 * HH + j] = xb;
    cat2[b * 2 * HH + j] = xb;
    cat1[b * 2 * HH + HH + j] = to_bf16(h[b * HH + j]);
  }
}

// softmax over S=64 attention scores per batch row
__global__ void __launch_bounds__(64)
softmax64(const float* __restrict__ scores, float* __restrict__ w) {
  __shared__ float sm[64];
  int b = blockIdx.x, t = threadIdx.x;
  float v = scores[b * SQ + t];
  sm[t] = v; __syncthreads();
  for (int s = 32; s > 0; s >>= 1) { if (t < s) sm[t] = fmaxf(sm[t], sm[t + s]); __syncthreads(); }
  float m = sm[0]; __syncthreads();
  float e = __expf(v - m);
  sm[t] = e; __syncthreads();
  for (int s = 32; s > 0; s >>= 1) { if (t < s) sm[t] += sm[t + s]; __syncthreads(); }
  w[b * SQ + t] = e / sm[0];
}

// attn_values[b,d] = sum_s w[b,s]*enc_out[s,b,d]  (batched weighting, not a GEMM)
__global__ void __launch_bounds__(256)
attn_values_kernel(const float* __restrict__ w, const float* __restrict__ enc_out,
                   __bf16* __restrict__ cat2) {
  int idx = blockIdx.x * 256 + threadIdx.x;   // 32*1024 threads
  int b = idx >> 10, d = idx & (HH - 1);
  float s = 0.0f;
#pragma unroll 4
  for (int t = 0; t < SQ; ++t)
    s += w[b * SQ + t] * enc_out[((size_t)t * BB + b) * HH + d];
  cat2[b * 2 * HH + HH + d] = to_bf16(s);
}

// LSTM gate nonlinearity: g[32,4096] (i|f|g|o) -> h,c ; also bf16 h, enc_out row
__global__ void __launch_bounds__(256)
lstm_gates(const float* __restrict__ g, float* __restrict__ h, float* __restrict__ c,
           __bf16* __restrict__ h_bf, float* __restrict__ enc_out_row) {
  int idx = blockIdx.x * 256 + threadIdx.x;   // 32*1024 threads
  int b = idx >> 10, j = idx & (HH - 1);
  const float* gr = g + (size_t)b * 4 * HH;
  float gi = gr[j];
  float gf = gr[HH + j];
  float gg = gr[2 * HH + j];
  float go = gr[3 * HH + j];
  float si = 1.0f / (1.0f + __expf(-gi));
  float sf = 1.0f / (1.0f + __expf(-gf));
  float so = 1.0f / (1.0f + __expf(-go));
  float cn = sf * c[idx] + si * tanhf(gg);
  float hn = so * tanhf(cn);
  c[idx] = cn;
  h[idx] = hn;
  h_bf[idx] = to_bf16(hn);
  if (enc_out_row) enc_out_row[idx] = hn;
}

// log-softmax over V=32000, NLL accumulation, argmax -> prev_tok
__global__ void __launch_bounds__(256)
loss_kernel(const float* __restrict__ logits, const int* __restrict__ target,
            int step, float* __restrict__ loss_out, int* __restrict__ prev_tok) {
  __shared__ float smx[256];
  __shared__ int   smi[256];
  int b = blockIdx.x, t = threadIdx.x;
  const float* row = logits + (size_t)b * VV;
  float m = -3.4e38f; int mi = 0;
  for (int j = t; j < VV; j += 256) {
    float v = row[j];
    if (v > m) { m = v; mi = j; }
  }
  smx[t] = m; smi[t] = mi;
  __syncthreads();
  for (int s = 128; s > 0; s >>= 1) {
    if (t < s && smx[t + s] > smx[t]) { smx[t] = smx[t + s]; smi[t] = smi[t + s]; }
    __syncthreads();
  }
  float gmax = smx[0];
  int gidx = smi[0];
  __syncthreads();
  float se = 0.0f;
  for (int j = t; j < VV; j += 256) se += __expf(row[j] - gmax);
  smx[t] = se;
  __syncthreads();
  for (int s = 128; s > 0; s >>= 1) {
    if (t < s) smx[t] += smx[t + s];
    __syncthreads();
  }
  if (t == 0) {
    int tgt = target[step * BB + b];
    float lp = row[tgt] - gmax - logf(smx[0]);
    atomicAdd(loss_out, -lp * (1.0f / (float)BB));
    prev_tok[b] = gidx;
  }
}

// ---------------------------------------------------------------------------
// Host orchestration (graph-capturable; all launches on `stream`)
// ---------------------------------------------------------------------------
extern "C" void kernel_launch(void* const* d_in, const int* in_sizes, int n_in,
                              void* d_out, int out_size, void* d_ws, size_t ws_size,
                              hipStream_t stream) {
  (void)in_sizes; (void)n_in; (void)out_size; (void)ws_size;

  const int*   input    = (const int*)d_in[0];
  const int*   target   = (const int*)d_in[1];
  const int*   use_tf   = (const int*)d_in[2];
  const float* enc_emb  = (const float*)d_in[3];
  const float* enc_Wih  = (const float*)d_in[4];
  const float* enc_Whh  = (const float*)d_in[5];
  const float* enc_b    = (const float*)d_in[6];
  const float* dec_emb  = (const float*)d_in[7];
  const float* dec_Wih  = (const float*)d_in[8];
  const float* dec_Whh  = (const float*)d_in[9];
  const float* dec_b    = (const float*)d_in[10];
  const float* attn_W   = (const float*)d_in[11];
  const float* attn_b   = (const float*)d_in[12];
  const float* comb_W   = (const float*)d_in[13];
  const float* comb_b   = (const float*)d_in[14];
  const float* out_W    = (const float*)d_in[15];
  const float* out_b    = (const float*)d_in[16];
  float* loss = (float*)d_out;

  // workspace carve-up (256B aligned)
  char* base = (char*)d_ws;
  size_t off = 0;
  auto carve = [&](size_t bytes) -> void* {
    void* p = base + off;
    off = (off + bytes + 255) & ~(size_t)255;
    return p;
  };
  __bf16* enc_Wih_bf = (__bf16*)carve((size_t)4 * HH * HH * 2);
  __bf16* enc_Whh_bf = (__bf16*)carve((size_t)4 * HH * HH * 2);
  __bf16* dec_Wih_bf = (__bf16*)carve((size_t)4 * HH * HH * 2);
  __bf16* dec_Whh_bf = (__bf16*)carve((size_t)4 * HH * HH * 2);
  __bf16* comb_W_bf  = (__bf16*)carve((size_t)HH * 2 * HH * 2);
  __bf16* attn_W_bf  = (__bf16*)carve((size_t)SQ * 2 * HH * 2);
  __bf16* out_W_bf   = (__bf16*)carve((size_t)VV * HH * 2);     // 65.5 MB, L2-resident
  float*  h_f        = (float*)carve((size_t)BB * HH * 4);
  float*  c_f        = (float*)carve((size_t)BB * HH * 4);
  __bf16* h_bf       = (__bf16*)carve((size_t)BB * HH * 2);
  __bf16* x_bf       = (__bf16*)carve((size_t)BB * HH * 2);
  float*  g_buf      = (float*)carve((size_t)BB * 4 * HH * 4);
  float*  enc_out    = (float*)carve((size_t)SQ * BB * HH * 4);
  float*  x_dec      = (float*)carve((size_t)BB * HH * 4);
  __bf16* cat1       = (__bf16*)carve((size_t)BB * 2 * HH * 2);
  __bf16* cat2       = (__bf16*)carve((size_t)BB * 2 * HH * 2);
  float*  scores     = (float*)carve((size_t)BB * SQ * 4);
  float*  attnw      = (float*)carve((size_t)BB * SQ * 4);
  float*  logits     = (float*)carve((size_t)BB * VV * 4);
  int*    prev_tok   = (int*)carve((size_t)BB * 4);

  // one-time bf16 packing of all weights (in-graph; deterministic per call)
  auto cvt = [&](const float* s, __bf16* d, int n) {
    cvt_bf16<<<dim3((n / 4 + 255) / 256), dim3(256), 0, stream>>>(s, d, n);
  };
  cvt(enc_Wih, enc_Wih_bf, 4 * HH * HH);
  cvt(enc_Whh, enc_Whh_bf, 4 * HH * HH);
  cvt(dec_Wih, dec_Wih_bf, 4 * HH * HH);
  cvt(dec_Whh, dec_Whh_bf, 4 * HH * HH);
  cvt(comb_W,  comb_W_bf,  HH * 2 * HH);
  cvt(attn_W,  attn_W_bf,  SQ * 2 * HH);
  cvt(out_W,   out_W_bf,   VV * HH);

  init_state<<<dim3(128), dim3(256), 0, stream>>>(h_f, c_f, h_bf, prev_tok, loss);

  // ---- encoder: 64 sequential LSTM steps ----
  for (int s = 0; s < SQ; ++s) {
    enc_prep<<<dim3(BB), dim3(256), 0, stream>>>(input, enc_emb, x_bf, s);
    // g = x*Wih^T + h*Whh^T + b : 256 wave-groups -> 32 blocks * 8 waves
    wmma_gemm_dual2<<<dim3(32), dim3(256), 0, stream>>>(x_bf, enc_Wih_bf, h_bf, enc_Whh_bf,
                                                        enc_b, g_buf, 4 * HH, HH);
    lstm_gates<<<dim3(128), dim3(256), 0, stream>>>(g_buf, h_f, c_f, h_bf,
                                                    enc_out + (size_t)s * BB * HH);
  }

  // ---- decoder with attention: 64 sequential steps ----
  for (int t = 0; t < TQ; ++t) {
    dec_prep<<<dim3(BB), dim3(256), 0, stream>>>(target, use_tf, prev_tok, dec_emb,
                                                 h_f, x_dec, cat1, cat2, t);
    // attention scores: [32,2048]x[2048,64] -> 2 wave-groups -> 1 block
    wmma_gemm4<<<dim3(1), dim3(256), 0, stream>>>(cat1, attn_W_bf, attn_b,
                                                  scores, SQ, 2 * HH);
    softmax64<<<dim3(BB), dim3(64), 0, stream>>>(scores, attnw);
    attn_values_kernel<<<dim3(128), dim3(256), 0, stream>>>(attnw, enc_out, cat2);
    // comb: relu([x|attn]*comb_W^T + b) -> bf16 : 32 wave-groups -> 4 blocks
    wmma_gemm4_relu_bf16<<<dim3(4), dim3(256), 0, stream>>>(cat2, comb_W_bf, comb_b,
                                                            x_bf, HH, 2 * HH);
    // decoder LSTM
    wmma_gemm_dual2<<<dim3(32), dim3(256), 0, stream>>>(x_bf, dec_Wih_bf, h_bf, dec_Whh_bf,
                                                        dec_b, g_buf, 4 * HH, HH);
    lstm_gates<<<dim3(128), dim3(256), 0, stream>>>(g_buf, h_f, c_f, h_bf, nullptr);
    // logits: [32,1024]x[1024,32000] -> 1000 wave-groups -> 125 blocks (out_W_bf in L2)
    wmma_gemm4<<<dim3(125), dim3(256), 0, stream>>>(h_bf, out_W_bf, out_b,
                                                    logits, VV, HH);
    loss_kernel<<<dim3(BB), dim3(256), 0, stream>>>(logits, target, t, loss, prev_tok);
  }
}
